// HaloBlock_60670708023313
// MI455X (gfx1250) — matrix operand order repaired and checked by reading, not compiled, not run on hardware
//
#include <hip/hip_runtime.h>
#include <hip/hip_bf16.h>
#include <math.h>
#include <stdint.h>

typedef __attribute__((ext_vector_type(16))) _Float16 v16h;
typedef __attribute__((ext_vector_type(8)))  float    v8f;

// ---------------------------------------------------------------------------
// WMMA helpers (CDNA5 v_wmma_f32_16x16x32_f16, wave32)
//   A: 16x32 (MxK)  lane L<16: row L, K {k0..k0+15}
//                   lane L>=16: row L-16, K {k0+16..k0+31}
//   B: 32x16 (KxN)  lane L<16: col L, K {k0..k0+15}; lane L>=16: col L-16, K {k0+16..k0+31}
//   C/D: 16x16 f32  lane L: col L&15, rows r + (L<16 ? 0 : 8)
// ---------------------------------------------------------------------------
__device__ inline v8f wmma_f16(v16h a, v16h b, v8f c) {
  return __builtin_amdgcn_wmma_f32_16x16x32_f16(false, a, false, b, (short)0, c,
                                                false, false);
}

__device__ inline v16h load_a_frag(const _Float16* __restrict__ A, int ld, int k0,
                                   int lane) {
  int r = lane & 15;
  int koff = (lane & 16) ? 16 : 0;
  const _Float16* p = A + (size_t)r * ld + k0 + koff;
  v16h a;
#pragma unroll
  for (int i = 0; i < 8; ++i) a[i] = p[i];
#pragma unroll
  for (int i = 0; i < 8; ++i) a[i + 8] = p[i + 8];
  return a;
}

// Weight-style operand: Wt is row-major (N x K); B(k,n) = Wt[n*ld + k]
__device__ inline v16h load_b_frag_nk(const _Float16* __restrict__ Wt, int ld,
                                      int k0, int lane) {
  int n = lane & 15;
  int kb = (lane & 16) ? 16 : 0;
  const _Float16* p = Wt + (size_t)n * ld + k0 + kb;
  v16h b;
#pragma unroll
  for (int i = 0; i < 16; ++i) b[i] = p[i];
  return b;
}

// V-style operand: Vm is row-major (K x N); B(k,n) = Vm[k*ld + n]
__device__ inline v16h load_b_frag_kn(const _Float16* __restrict__ Vm, int ld,
                                      int n0, int k0, int lane) {
  int n = n0 + (lane & 15);
  int kb = k0 + ((lane & 16) ? 16 : 0);
  v16h b;
#pragma unroll
  for (int i = 0; i < 16; ++i) b[i] = Vm[(size_t)(kb + i) * ld + n];
  return b;
}

// Async global -> LDS copy of 16 bytes (GLOBAL_LOAD_ASYNC_TO_LDS_B128, ASYNCcnt)
__device__ inline void async_copy_b128(uint32_t lds_addr, const void* gaddr) {
  asm volatile("global_load_async_to_lds_b128 %0, %1, off"
               :
               : "v"(lds_addr), "v"((uint64_t)(uintptr_t)gaddr)
               : "memory");
}

__device__ inline void wait_async0() {
#if __has_builtin(__builtin_amdgcn_s_wait_asynccnt)
  __builtin_amdgcn_s_wait_asynccnt(0);
#else
  asm volatile("s_wait_asynccnt 0x0" ::: "memory");
#endif
}

// ---------------------------------------------------------------------------
// f32 -> f16 weight conversion
// ---------------------------------------------------------------------------
__global__ void cvt_f16_kernel(const float* __restrict__ src,
                               _Float16* __restrict__ dst, int n) {
  int i = blockIdx.x * blockDim.x + threadIdx.x;
  if (i < n) dst[i] = (_Float16)src[i];
}

// ---------------------------------------------------------------------------
// Channels-first LayerNorm over C=256: NCHW f32 -> NHWC f16
// one pixel per 256-thread block
// ---------------------------------------------------------------------------
__global__ __launch_bounds__(256)
void ln_kernel(const float* __restrict__ x, const float* __restrict__ w,
               const float* __restrict__ b, _Float16* __restrict__ out) {
  int p = blockIdx.x;               // 0 .. B*H*W-1
  int bimg = p / 3136;
  int hw = p % 3136;
  int c = threadIdx.x;
  float v = x[((size_t)bimg * 256 + c) * 3136 + hw];
  float s = v, s2 = v * v;
#pragma unroll
  for (int m = 16; m >= 1; m >>= 1) {
    s  += __shfl_xor(s, m, 32);
    s2 += __shfl_xor(s2, m, 32);
  }
  __shared__ float bs[8], bs2[8];
  int wv = threadIdx.x >> 5, ln = threadIdx.x & 31;
  if (ln == 0) { bs[wv] = s; bs2[wv] = s2; }
  __syncthreads();
  s = 0.f; s2 = 0.f;
#pragma unroll
  for (int i = 0; i < 8; ++i) { s += bs[i]; s2 += bs2[i]; }
  float mean = s * (1.0f / 256.0f);
  float var = s2 * (1.0f / 256.0f) - mean * mean;
  float xn = (v - mean) * rsqrtf(var + 1e-6f);
  out[(size_t)p * 256 + c] = (_Float16)(xn * w[c] + b[c]);
}

// ---------------------------------------------------------------------------
// Generic WMMA GEMM: C = epilogue(A(MxK,f16) @ Wt(NxK,f16)^T)
// block = 128 threads (4 waves); workgroup tile 64x64, each wave 16x64.
// Weight tile (64x32 per k-step) is staged to LDS with async global->LDS
// copies, double-buffered one k-step ahead of the WMMA consumption.
// MODE 0: Ch = acc*scale (f16)
// MODE 1: Ch = gelu(acc + bias) (f16, exact erf GELU)
// MODE 2: Cf[NCHW] = resid[NCHW] + gamma[n]*(acc + bias[n])   (N must be 256)
// ---------------------------------------------------------------------------
template <int MODE>
__global__ __launch_bounds__(128)
void gemm_kernel(const _Float16* __restrict__ A, const _Float16* __restrict__ Wt,
                 const float* __restrict__ bias, _Float16* __restrict__ Ch,
                 const float* __restrict__ resid, const float* __restrict__ gamma,
                 float* __restrict__ Cf, int M, int N, int K, float scale) {
  __shared__ alignas(16) _Float16 bstage[2][64 * 32];  // 8 KB double buffer

  int tid = threadIdx.x;
  int lane = tid & 31;
  int wv = tid >> 5;
  int m0 = blockIdx.x * 64 + wv * 16;
  int n0 = blockIdx.y * 64;
  v8f zero = {};
  v8f acc[4];
#pragma unroll
  for (int t = 0; t < 4; ++t) acc[t] = zero;

  const _Float16* Arow = A + (size_t)m0 * K;

  // stage one 64x32 weight tile (4 KB) into bstage[buf]: 256 x 16B chunks
  auto stage = [&](int k0, int buf) {
#pragma unroll
    for (int s = 0; s < 2; ++s) {
      int idx = tid + s * 128;          // 0..255
      int n = idx >> 2, ch = idx & 3;
      const _Float16* gp = Wt + (size_t)(n0 + n) * K + k0 + ch * 8;
      uint32_t lds = (uint32_t)(uintptr_t)(&bstage[buf][n * 32 + ch * 8]);
      async_copy_b128(lds, gp);
    }
  };

  stage(0, 0);
  wait_async0();
  __syncthreads();

  int cur = 0;
  for (int k0 = 0; k0 < K; k0 += 32) {
    if (k0 + 32 < K) stage(k0 + 32, cur ^ 1);          // prefetch next tile
    __builtin_prefetch(Arow + (size_t)(lane & 15) * K + k0 + 64, 0, 1);
    v16h af = load_a_frag(Arow, K, k0, lane);
#pragma unroll
    for (int t = 0; t < 4; ++t) {
      v16h bf = load_b_frag_nk(&bstage[cur][t * 16 * 32], 32, 0, lane);
      acc[t] = wmma_f16(af, bf, acc[t]);
    }
    wait_async0();
    __syncthreads();
    cur ^= 1;
  }

  int half = (lane & 16) ? 8 : 0;
  int nlo = lane & 15;
#pragma unroll
  for (int t = 0; t < 4; ++t) {
    int n = n0 + t * 16 + nlo;
#pragma unroll
    for (int r = 0; r < 8; ++r) {
      int m = m0 + r + half;
      float v = acc[t][r];
      if (MODE == 0) {
        Ch[(size_t)m * N + n] = (_Float16)(v * scale);
      } else if (MODE == 1) {
        float u = v + bias[n];
        float g = 0.5f * u * (1.0f + erff(u * 0.70710678118f));
        Ch[(size_t)m * N + n] = (_Float16)g;
      } else {
        float u = v + bias[n];
        int bimg = m / 3136, hw = m % 3136;
        size_t idx = ((size_t)bimg * 256 + n) * 3136 + hw;
        Cf[idx] = resid[idx] + gamma[n] * u;
      }
    }
  }
}

// ---------------------------------------------------------------------------
// Halo attention: one block per (window, head). 128 threads (4 waves).
// Q is pre-scaled by dhead^-0.5 (folded into Q projection).
// ---------------------------------------------------------------------------
__global__ __launch_bounds__(128)
void halo_attn_kernel(const _Float16* __restrict__ Qp,
                      const _Float16* __restrict__ KVp,
                      const float* __restrict__ rel_h,
                      const float* __restrict__ rel_w,
                      _Float16* __restrict__ Yp) {
  __shared__ _Float16 q_s[64 * 32];    // queries, rows 49..63 zero
  __shared__ _Float16 k_s[192 * 32];   // keys, rows 169..191 zero
  __shared__ _Float16 v_s[192 * 32];   // values
  __shared__ _Float16 p_s[64 * 192];   // attention probs (f16)
  __shared__ float rw_s[49 * 13];      // q . rel_w[j - y + 12]
  __shared__ float rh_s[49 * 13];      // q . rel_h[i - x + 12]

  int win = blockIdx.x;
  int head = blockIdx.y;
  int b = win >> 6;
  int wy = (win >> 3) & 7;
  int wx = win & 7;
  int tid = threadIdx.x;
  int lane = tid & 31;
  int wv = tid >> 5;

  for (int idx = tid; idx < 64 * 32; idx += 128) {
    int row = idx >> 5, c = idx & 31;
    _Float16 val = (_Float16)0.0f;
    if (row < 49) {
      int qx = row / 7, qy = row % 7;
      int h = wy * 7 + qx, w = wx * 7 + qy;
      val = Qp[(((size_t)b * 56 + h) * 56 + w) * 256 + head * 32 + c];
    }
    q_s[idx] = val;
  }
  for (int idx = tid; idx < 192 * 32; idx += 128) {
    int row = idx >> 5, c = idx & 31;
    _Float16 kk = (_Float16)0.0f, vv = (_Float16)0.0f;
    if (row < 169) {
      int i = row / 13, j = row % 13;
      int h = wy * 7 + i - 3, w = wx * 7 + j - 3;
      if (h >= 0 && h < 56 && w >= 0 && w < 56) {
        size_t base = (((size_t)b * 56 + h) * 56 + w) * 512 + head * 32 + c;
        kk = KVp[base];
        vv = KVp[base + 256];
      }
    }
    k_s[idx] = kk;
    v_s[idx] = vv;
  }
  for (int idx = tid; idx < 64 * 192; idx += 128) p_s[idx] = (_Float16)0.0f;
  __syncthreads();

  // content-dependent relative position logits (shared across the window)
  for (int idx = tid; idx < 49 * 13; idx += 128) {
    int q = idx / 13, p = idx % 13;
    int qx = q / 7, qy = q % 7;
    const float* rwv = rel_w + (p - qy + 12) * 32;
    const float* rhv = rel_h + (p - qx + 12) * 32;
    float sw = 0.f, sh = 0.f;
#pragma unroll
    for (int c = 0; c < 32; ++c) {
      float qq = (float)q_s[q * 32 + c];
      sw += qq * rwv[c];
      sh += qq * rhv[c];
    }
    rw_s[idx] = sw;
    rh_s[idx] = sh;
  }
  __syncthreads();

  // logits: Q(64x32) @ K^T(32x176) -> each wave 16 rows x 11 tiles, K=32 exact
  v8f zero = {};
  v8f sim[11];
#pragma unroll
  for (int t = 0; t < 11; ++t) sim[t] = zero;
  {
    v16h af = load_a_frag(q_s + wv * 16 * 32, 32, 0, lane);
#pragma unroll
    for (int t = 0; t < 11; ++t) {
      v16h bf = load_b_frag_nk(k_s + t * 16 * 32, 32, 0, lane);
      sim[t] = wmma_f16(af, bf, sim[t]);
    }
  }

  int half = (lane & 16) ? 8 : 0;
  int nlo = lane & 15;
  for (int r = 0; r < 8; ++r) {
    int m = wv * 16 + r + half;  // query row (0..63)
    float vals[11];
    float mx = -1e30f;
#pragma unroll
    for (int t = 0; t < 11; ++t) {
      int col = t * 16 + nlo;
      float v = -1e30f;
      if (col < 169) {
        int i = col / 13, j = col % 13;
        int h = wy * 7 + i - 3, w = wx * 7 + j - 3;
        if (h >= 0 && h < 56 && w >= 0 && w < 56) {
          v = sim[t][r];
          if (m < 49) v += rw_s[m * 13 + j] + rh_s[m * 13 + i];
        }
      }
      vals[t] = v;
      mx = fmaxf(mx, v);
    }
#pragma unroll
    for (int s = 1; s < 16; s <<= 1) mx = fmaxf(mx, __shfl_xor(mx, s, 16));
    float sum = 0.f;
#pragma unroll
    for (int t = 0; t < 11; ++t) {
      float e = __expf(vals[t] - mx);
      vals[t] = e;
      sum += e;
    }
#pragma unroll
    for (int s = 1; s < 16; s <<= 1) sum += __shfl_xor(sum, s, 16);
    float inv = 1.0f / sum;
#pragma unroll
    for (int t = 0; t < 11; ++t)
      p_s[m * 192 + t * 16 + nlo] = (_Float16)(vals[t] * inv);
  }
  __syncthreads();

  // out = P(64x192) @ V(192x32); 6 k-steps, 2 n-tiles per wave
  v8f oacc[2] = {zero, zero};
  for (int ks = 0; ks < 6; ++ks) {
    v16h pa = load_a_frag(p_s + wv * 16 * 192, 192, ks * 32, lane);
#pragma unroll
    for (int t = 0; t < 2; ++t) {
      v16h bv = load_b_frag_kn(v_s, 32, t * 16, ks * 32, lane);
      oacc[t] = wmma_f16(pa, bv, oacc[t]);
    }
  }
#pragma unroll
  for (int t = 0; t < 2; ++t) {
    int n = t * 16 + nlo;
#pragma unroll
    for (int r = 0; r < 8; ++r) {
      int m = wv * 16 + r + half;
      if (m < 49) {
        int qx = m / 7, qy = m % 7;
        int h = wy * 7 + qx, w = wx * 7 + qy;
        Yp[(((size_t)b * 56 + h) * 56 + w) * 256 + head * 32 + n] =
            (_Float16)oacc[t][r];
      }
    }
  }
}

// ---------------------------------------------------------------------------
// Host-side orchestration
// ---------------------------------------------------------------------------
extern "C" void kernel_launch(void* const* d_in, const int* in_sizes, int n_in,
                              void* d_out, int out_size, void* d_ws,
                              size_t ws_size, hipStream_t stream) {
  const float* x      = (const float*)d_in[0];
  const float* ln1_w  = (const float*)d_in[1];
  const float* ln1_b  = (const float*)d_in[2];
  const float* ln2_w  = (const float*)d_in[3];
  const float* ln2_b  = (const float*)d_in[4];
  const float* wq     = (const float*)d_in[5];
  const float* wkv    = (const float*)d_in[6];
  const float* wo     = (const float*)d_in[7];
  const float* bo     = (const float*)d_in[8];
  const float* rel_h  = (const float*)d_in[9];
  const float* rel_w  = (const float*)d_in[10];
  const float* gamma1 = (const float*)d_in[11];
  const float* gamma2 = (const float*)d_in[12];
  const float* fc1_w  = (const float*)d_in[13];
  const float* fc1_b  = (const float*)d_in[14];
  const float* fc2_w  = (const float*)d_in[15];
  const float* fc2_b  = (const float*)d_in[16];

  const int TOK = 16 * 56 * 56;  // 50176 tokens

  // workspace layout (f16 unless noted)
  _Float16* wq_h  = (_Float16*)d_ws;                    //  256*256
  _Float16* wkv_h = wq_h  + 256 * 256;                  //  512*256
  _Float16* wo_h  = wkv_h + 512 * 256;                  //  256*256
  _Float16* fc1_h = wo_h  + 256 * 256;                  // 1024*256
  _Float16* fc2_h = fc1_h + 1024 * 256;                 //  256*1024
  _Float16* xn_h  = fc2_h + 256 * 1024;                 // TOK*256 (LN1 out; reused for LN2 out)
  _Float16* big   = xn_h + (size_t)TOK * 256;           // TOK*1024 (Q|KV|Y; reused for MLP hidden)
  _Float16* Qp    = big;                                // TOK*256
  _Float16* KVp   = Qp + (size_t)TOK * 256;             // TOK*512
  _Float16* Yp    = KVp + (size_t)TOK * 512;            // TOK*256
  float*    x1    = (float*)(big + (size_t)TOK * 1024); // TOK*256 f32 NCHW

  // 1) weights -> f16
  cvt_f16_kernel<<<(256 * 256 + 255) / 256, 256, 0, stream>>>(wq, wq_h, 256 * 256);
  cvt_f16_kernel<<<(512 * 256 + 255) / 256, 256, 0, stream>>>(wkv, wkv_h, 512 * 256);
  cvt_f16_kernel<<<(256 * 256 + 255) / 256, 256, 0, stream>>>(wo, wo_h, 256 * 256);
  cvt_f16_kernel<<<(1024 * 256 + 255) / 256, 256, 0, stream>>>(fc1_w, fc1_h, 1024 * 256);
  cvt_f16_kernel<<<(256 * 1024 + 255) / 256, 256, 0, stream>>>(fc2_w, fc2_h, 256 * 1024);

  dim3 blk(128);

  // 2) LN1 -> f16 NHWC
  ln_kernel<<<TOK, 256, 0, stream>>>(x, ln1_w, ln1_b, xn_h);

  // 3) Q projection (dhead^-0.5 folded in) and KV projection
  gemm_kernel<0><<<dim3(TOK / 64, 4), blk, 0, stream>>>(
      xn_h, wq_h, nullptr, Qp, nullptr, nullptr, nullptr, TOK, 256, 256,
      0.17677669529663687f);
  gemm_kernel<0><<<dim3(TOK / 64, 8), blk, 0, stream>>>(
      xn_h, wkv_h, nullptr, KVp, nullptr, nullptr, nullptr, TOK, 512, 256, 1.0f);

  // 4) halo attention: 1024 windows x 8 heads
  halo_attn_kernel<<<dim3(1024, 8), blk, 0, stream>>>(Qp, KVp, rel_h, rel_w, Yp);

  // 5) out projection + bias + gamma1 residual -> x1 (NCHW f32)
  gemm_kernel<2><<<dim3(TOK / 64, 4), blk, 0, stream>>>(
      Yp, wo_h, bo, nullptr, x, gamma1, x1, TOK, 256, 256, 1.0f);

  // 6) LN2 -> f16 NHWC (reuse xn_h)
  ln_kernel<<<TOK, 256, 0, stream>>>(x1, ln2_w, ln2_b, xn_h);

  // 7) fc1 + bias + exact GELU -> hidden (reuse big)
  gemm_kernel<1><<<dim3(TOK / 64, 16), blk, 0, stream>>>(
      xn_h, fc1_h, fc1_b, big, nullptr, nullptr, nullptr, TOK, 1024, 256, 1.0f);

  // 8) fc2 + bias + gamma2 residual -> d_out (NCHW f32)
  gemm_kernel<2><<<dim3(TOK / 64, 4), blk, 0, stream>>>(
      big, fc2_h, fc2_b, nullptr, x1, gamma2, (float*)d_out, TOK, 256, 1024, 1.0f);
}
